// CDE_STGNN_59897613910629
// MI455X (gfx1250) — matrix operand order repaired and checked by reading, not compiled, not run on hardware
//
#include <hip/hip_runtime.h>
#include <hip/hip_bf16.h>
#include <math.h>

// Problem dims (fixed by the reference)
#define B_TOT   256
#define T_KNOTS 8
#define D_IN    289
#define H_DIM   64
#define Z_OUT   10
#define NB      16            // batch rows per workgroup (WMMA N tile)
#define DTILES  19            // ceil(289/16)
#define DPAD    (DTILES * 16) // 304, zero-padded D
#define NTILE   (H_DIM * DTILES) // 1216 A-tiles of 16 rows each

typedef __attribute__((ext_vector_type(16))) __bf16        v16bf;
typedef __attribute__((ext_vector_type(16))) unsigned short v16u;
typedef __attribute__((ext_vector_type(8)))  float          v8f;

#if defined(__has_builtin)
#if __has_builtin(__builtin_amdgcn_tanhf)
#define TANHF(x) __builtin_amdgcn_tanhf(x)
#else
#define TANHF(x) tanhf(x)
#endif
#else
#define TANHF(x) tanhf(x)
#endif

__device__ __forceinline__ unsigned short f2bf_bits(float f) {
    unsigned u = __builtin_bit_cast(unsigned, f);
    unsigned r = u + 0x7FFFu + ((u >> 16) & 1u);   // round-to-nearest-even
    return (unsigned short)(r >> 16);
}

// ---------------------------------------------------------------------------
// Prep 1: repack W_f (fp32 [18496,64]) into bf16 WMMA-A-fragment order.
// Tile t = h_out*19 + dtile covers W_f rows h_out*289 + dtile*16 + M, M=0..15
// (zero-padded past d=288). Fragment layout per ISA 7.12.2 (16-bit A 16x32):
//   lane<16 : M=lane,    K = kt*32 +     (e + (e>=8 ? 8:0))
//   lane>=16: M=lane-16, K = kt*32 + 8 + (e + (e>=8 ? 8:0))
// Flat u16 index: ((t*2 + kt)*32 + lane)*16 + e  -> one 32B load per lane.
// Also emits padded bias bpad[t*16 + M].
// ---------------------------------------------------------------------------
__global__ void cde_pack_wf(const float* __restrict__ Wf,
                            const float* __restrict__ bf,
                            unsigned short* __restrict__ Wpk,
                            float* __restrict__ bpad) {
    int idx = blockIdx.x * blockDim.x + threadIdx.x;
    if (idx < NTILE * 1024) {
        int e    = idx & 15;
        int lane = (idx >> 4) & 31;
        int kt   = (idx >> 9) & 1;
        int t    = idx >> 10;
        int h_out = t / DTILES;
        int dtile = t % DTILES;
        int half  = lane >> 4;
        int M     = lane & 15;
        int K     = kt * 32 + half * 8 + e + ((e >= 8) ? 8 : 0);
        int d     = dtile * 16 + M;
        float v = 0.0f;
        if (d < D_IN) v = Wf[(h_out * D_IN + d) * H_DIM + K];
        Wpk[idx] = f2bf_bits(v);
    }
    if (idx < NTILE * 16) {
        int M = idx & 15;
        int t = idx >> 4;
        int h_out = t / DTILES;
        int d     = (t % DTILES) * 16 + M;
        bpad[idx] = (d < D_IN) ? bf[h_out * D_IN + d] : 0.0f;
    }
}

// ---------------------------------------------------------------------------
// Prep 2: Hermite-cubic control-path derivatives g(b, interval, s, d) at the
// 5 normalized positions s = {0,.25,.5,.75,1} used by 2 RK4 substeps.
// Layout [B][7][5][DPAD] fp32, zero-padded in d (kills padded-row tanh terms).
// ---------------------------------------------------------------------------
__global__ void cde_prep_g(const float* __restrict__ times,
                           const float* __restrict__ X,
                           float* __restrict__ G) {
    int idx = blockIdx.x * blockDim.x + threadIdx.x;
    const int total = B_TOT * 7 * 5 * DPAD;
    if (idx >= total) return;
    int dd = idx % DPAD;
    int r  = idx / DPAD;
    int s5 = r % 5;  r /= 5;
    int i  = r % 7;
    int b  = r / 7;
    float g = 0.0f;
    if (dd < D_IN) {
        float dti = times[i + 1] - times[i];
        float Xi  = X[(b * T_KNOTS + i)     * D_IN + dd];
        float Xip = X[(b * T_KNOTS + i + 1) * D_IN + dd];
        float diffi = (Xip - Xi) / dti;
        float di;
        if (i == 0) {
            di = diffi;
        } else {
            float dtm = times[i] - times[i - 1];
            di = (Xi - X[(b * T_KNOTS + i - 1) * D_IN + dd]) / dtm;
        }
        float dip = diffi;
        float s  = 0.25f * (float)s5;
        float s2 = s * s;
        g = (Xip - Xi) * ((6.0f * s - 6.0f * s2) / dti)
          + di  * (3.0f * s2 - 4.0f * s + 1.0f)
          + dip * (3.0f * s2 - 2.0f * s);
    }
    G[idx] = g;
}

// ---------------------------------------------------------------------------
// Main: one workgroup per 16-batch tile, 8 waves (wave32). Each wave owns
// h_out = wave*8..+7. Per RK4 stage: build B fragments (h + c*k_prev -> bf16),
// then per (h_out, dtile): 2x v_wmma_f32_16x16x32_bf16, tanh+bias, weighted
// accumulate with g; half-lane shfl reduce -> dh. LDS keeps h/ktmp/kacc.
// launch_bounds(256,2): cap VGPRs <= 256/wave (no s_set_vgpr_msb churn) and
// run 4 waves/SIMD so occupancy hides the streamed A-fragment load latency.
// ---------------------------------------------------------------------------
__launch_bounds__(256, 2)
__global__ void cde_main(const float* __restrict__ times,
                         const float* __restrict__ z0,
                         const float* __restrict__ Wp,
                         const float* __restrict__ bp,
                         const unsigned short* __restrict__ Wpk,
                         const float* __restrict__ bpad,
                         const float* __restrict__ G,
                         float* __restrict__ out) {
    __shared__ float h[NB][H_DIM];
    __shared__ float ktmp[NB][H_DIM];
    __shared__ float kacc[NB][H_DIM];

    const int tid   = threadIdx.x;
    const int lane  = tid & 31;
    const int wave  = tid >> 5;
    const int half  = lane >> 4;
    const int n     = lane & 15;               // batch row within tile
    const int bbase = blockIdx.x * NB;

    const v16u*   WpkV  = (const v16u*)Wpk;
    const float4* bias4 = (const float4*)bpad;
    const float4* G4    = (const float4*)G;

    // h <- z0
    for (int x = tid; x < NB * H_DIM; x += 256)
        h[x >> 6][x & 63] = z0[(bbase + (x >> 6)) * H_DIM + (x & 63)];
    __syncthreads();

    auto project = [&](int tpt) {
        if (tid < NB * Z_OUT) {
            int pn = tid / Z_OUT, z = tid % Z_OUT;
            float acc = bp[z];
            #pragma unroll 8
            for (int k = 0; k < H_DIM; ++k) acc += h[pn][k] * Wp[z * H_DIM + k];
            out[((bbase + pn) * T_KNOTS + tpt) * Z_OUT + z] = acc;
        }
    };
    project(0);

    for (int iv = 0; iv < T_KNOTS - 1; ++iv) {
        float dti   = times[iv + 1] - times[iv];
        float hstep = dti * 0.5f;                       // dti / SUBSTEPS
        for (int kk = 0; kk < 2; ++kk) {
            #pragma unroll
            for (int st = 0; st < 4; ++st) {
                const int   sidx = 2 * kk + ((st == 0) ? 0 : (st == 3) ? 2 : 1);
                const float c    = (st == 0) ? 0.0f : ((st == 3) ? hstep : 0.5f * hstep);
                const float wgt  = (st == 1 || st == 2) ? 2.0f : 1.0f;

                // ---- build B fragments: h_stage = h + c * k_prev, bf16 ----
                // B (32x16, K x N): K = 8*half + e + (e>=8 ? 8:0); N=lane%16
                v16u bu0, bu1;
                const int koff = half * 8;
                #pragma unroll
                for (int e = 0; e < 16; ++e) {
                    const int ke = koff + e + ((e >= 8) ? 8 : 0);
                    float v0 = h[n][ke];
                    float v1 = h[n][32 + ke];
                    if (st > 0) {
                        v0 += c * ktmp[n][ke];
                        v1 += c * ktmp[n][32 + ke];
                    }
                    bu0[e] = f2bf_bits(v0);
                    bu1[e] = f2bf_bits(v1);
                }
                const v16bf bfr0 = __builtin_bit_cast(v16bf, bu0);
                const v16bf bfr1 = __builtin_bit_cast(v16bf, bu1);
                __syncthreads();   // everyone done reading h/ktmp

                const int gbase = (((bbase + n) * 7 + iv) * 5 + sidx) * DPAD;

                #pragma unroll 1
                for (int hh = 0; hh < 8; ++hh) {
                    const int h_out = wave * 8 + hh;
                    float partial = 0.0f;
                    #pragma unroll 1
                    for (int dtile = 0; dtile < DTILES; ++dtile) {
                        const int t = h_out * DTILES + dtile;
                        // Streamed once per stage, no L0 reuse -> non-temporal
                        v16u au0 = __builtin_nontemporal_load(&WpkV[(t * 2 + 0) * 32 + lane]);
                        v16u au1 = __builtin_nontemporal_load(&WpkV[(t * 2 + 1) * 32 + lane]);
                        v16bf a0 = __builtin_bit_cast(v16bf, au0);
                        v16bf a1 = __builtin_bit_cast(v16bf, au1);
                        v8f cacc = {};
                        cacc = __builtin_amdgcn_wmma_f32_16x16x32_bf16(
                                   false, a0, false, bfr0, (short)0, cacc, false, false);
                        cacc = __builtin_amdgcn_wmma_f32_16x16x32_bf16(
                                   false, a1, false, bfr1, (short)0, cacc, false, false);
                        // C[r] holds row M = r + 8*half of this tile, col N = n
                        float4 bl = bias4[t * 4 + half * 2];
                        float4 bh = bias4[t * 4 + half * 2 + 1];
                        int    gi = (gbase + dtile * 16 + half * 8) >> 2;
                        float4 gl = G4[gi];
                        float4 gh = G4[gi + 1];
                        partial += TANHF(cacc[0] + bl.x) * gl.x;
                        partial += TANHF(cacc[1] + bl.y) * gl.y;
                        partial += TANHF(cacc[2] + bl.z) * gl.z;
                        partial += TANHF(cacc[3] + bl.w) * gl.w;
                        partial += TANHF(cacc[4] + bh.x) * gh.x;
                        partial += TANHF(cacc[5] + bh.y) * gh.y;
                        partial += TANHF(cacc[6] + bh.z) * gh.z;
                        partial += TANHF(cacc[7] + bh.w) * gh.w;
                    }
                    partial += __shfl_xor(partial, 16, 32);  // fold lane halves
                    if (half == 0) {
                        ktmp[n][h_out] = partial;            // k_s for next stage
                        if (st == 0) kacc[n][h_out] = partial;
                        else         kacc[n][h_out] += wgt * partial;
                    }
                }
                __syncthreads();   // stage writes visible before next build
            }
            // h += (hstep/6) * (k1 + 2k2 + 2k3 + k4)
            const float w6 = hstep * (1.0f / 6.0f);
            for (int x = tid; x < NB * H_DIM; x += 256)
                h[x >> 6][x & 63] += w6 * kacc[x >> 6][x & 63];
            __syncthreads();
        }
        project(iv + 1);
    }
}

// ---------------------------------------------------------------------------
extern "C" void kernel_launch(void* const* d_in, const int* in_sizes, int n_in,
                              void* d_out, int out_size, void* d_ws, size_t ws_size,
                              hipStream_t stream) {
    (void)in_sizes; (void)n_in; (void)out_size; (void)ws_size;
    const float* times = (const float*)d_in[0];
    const float* X     = (const float*)d_in[1];
    const float* z0    = (const float*)d_in[2];
    const float* Wf    = (const float*)d_in[3];
    const float* bf    = (const float*)d_in[4];
    const float* Wp    = (const float*)d_in[5];
    const float* bp    = (const float*)d_in[6];
    float* out = (float*)d_out;

    char* ws = (char*)d_ws;
    unsigned short* Wpk = (unsigned short*)ws;                       // 2,490,368 B
    size_t off = (size_t)NTILE * 1024 * sizeof(unsigned short);
    float* bpad = (float*)(ws + off);                                // 77,824 B
    off += (size_t)NTILE * 16 * sizeof(float);
    float* G = (float*)(ws + off);                                   // 10,895,360 B

    const int npack = NTILE * 1024;
    cde_pack_wf<<<(npack + 255) / 256, 256, 0, stream>>>(Wf, bf, Wpk, bpad);

    const int ng = B_TOT * 7 * 5 * DPAD;
    cde_prep_g<<<(ng + 255) / 256, 256, 0, stream>>>(times, X, G);

    cde_main<<<B_TOT / NB, 256, 0, stream>>>(times, z0, Wp, bp, Wpk, bpad, G, out);
}
